// GCN_23940147708136
// MI455X (gfx1250) — matrix operand order repaired and checked by reading, not compiled, not run on hardware
//
#include <hip/hip_runtime.h>

// ---------------------------------------------------------------------------
// 2-layer GCN (GCNConv w/ self-loops + sym norm) for MI455X / gfx1250.
//  - GEMMs: LDS-staged f16 WMMA (v_wmma_f32_16x16x32_f16, f32 accumulate).
//    One block computes a 16-row strip across all column tiles; A staged once
//    per block, B staged transposed so fragments are contiguous ds_load_b128.
//  - Edge aggregation: wave-per-edge float4 gather + global_atomic_add_f32,
//    L2-resident by construction (xw/agg = 51.2MB each << 192MB L2).
// ---------------------------------------------------------------------------

typedef __attribute__((ext_vector_type(16))) _Float16 v16h;
typedef __attribute__((ext_vector_type(8)))  _Float16 v8h;
typedef __attribute__((ext_vector_type(8)))  float    v8f;

namespace {
constexpr int kN = 100000;   // nodes
constexpr int kE = 1600000;  // edges (excluding self loops)
constexpr int kF = 100;      // input features
constexpr int kH = 128;      // hidden
constexpr int kC = 47;       // classes
}

// ---------------- degree / normalization ----------------------------------

__global__ void deg_init_k(float* __restrict__ deg) {
  int i = blockIdx.x * blockDim.x + threadIdx.x;
  if (i < kN) deg[i] = 1.0f;  // self-loop contributes 1 to every node's degree
}

__global__ void deg_count_k(const int* __restrict__ dst, float* __restrict__ deg) {
  int i = blockIdx.x * blockDim.x + threadIdx.x;
  if (i < kE) atomicAdd(&deg[dst[i]], 1.0f);
}

__global__ void dinv_k(float* __restrict__ deg) {
  int i = blockIdx.x * blockDim.x + threadIdx.x;
  if (i < kN) {
    float d = deg[i];
    deg[i] = (d > 0.0f) ? rsqrtf(d) : 0.0f;
  }
}

// ---------------- WMMA GEMM: C[M,NREAL] = A[M,K] * B[K,NREAL] --------------
// grid.x = M/16 (M multiple of 16); blockDim = NTILECOL*32.
// Block b computes rows [16b, 16b+16) x all NREAL columns.
// LDS: Bt[n][k] f16 (transposed, padded stride) + Asm[r][k] f16.
// Fragment layouts as loaded here (wave32; A and B use the same k order):
//   A 16x32: lane L holds M=L%16; e   -> K=k0+16*(L/16)+e  (contiguous).
//   B 32x16: lane L holds N=L%16; e   -> K=k0+16*(L/16)+e  (contiguous!).
//   C/D f32: VGPR r -> M=r+8*(L/16), N=L%16.

template <int K, int NREAL, int NTILECOL>
__global__ __launch_bounds__(NTILECOL * 32) void wmma_gemm_k(
    const float* __restrict__ A, const float* __restrict__ B,
    float* __restrict__ C) {
  constexpr int KPAD = (K + 31) & ~31;
  constexpr int NPAD = NTILECOL * 16;
  constexpr int LDB  = KPAD + 8;  // f16 stride; 272B breaks 256B bank period
  constexpr int T    = NTILECOL * 32;

  __shared__ alignas(32) _Float16 Bt[NPAD * LDB];   // B transposed, f16
  __shared__ alignas(32) _Float16 Asm[16 * LDB];    // 16-row A strip, f16

  const int tid = threadIdx.x;
  const int tm  = blockIdx.x;

  // Stage B transposed (coalesced global reads along n).
  for (int i = tid; i < KPAD * NPAD; i += T) {
    const int k = i / NPAD;
    const int n = i - k * NPAD;
    float v = ((k < K) && (n < NREAL)) ? B[(size_t)k * NREAL + n] : 0.0f;
    Bt[n * LDB + k] = (_Float16)v;
  }
  // Stage the 16-row A strip (coalesced along k).
  for (int i = tid; i < 16 * KPAD; i += T) {
    const int r = i / KPAD;
    const int k = i - r * KPAD;
    float v = (k < K) ? A[(size_t)(tm * 16 + r) * K + k] : 0.0f;
    Asm[r * LDB + k] = (_Float16)v;
  }
  __syncthreads();

  const int lane = tid & 31;
  const int tn   = tid >> 5;       // column tile of this wave
  const int half = lane >> 4;      // 0: lanes 0-15, 1: lanes 16-31
  const int l15  = lane & 15;

  v8f acc = {};
#pragma unroll
  for (int k0 = 0; k0 < KPAD; k0 += 32) {
    // A fragment: two contiguous 8-f16 chunks (16B-aligned LDS loads).
    const _Float16* ap = &Asm[l15 * LDB + k0 + half * 16];
    v8h alo = *(const v8h*)ap;
    v8h ahi = *(const v8h*)(ap + 8);
    v16h a = __builtin_shufflevector(alo, ahi, 0, 1, 2, 3, 4, 5, 6, 7,
                                     8, 9, 10, 11, 12, 13, 14, 15);
    // B fragment: 16 contiguous f16 from the transposed tile.
    const _Float16* bp = &Bt[(tn * 16 + l15) * LDB + k0 + half * 16];
    v8h blo = *(const v8h*)bp;
    v8h bhi = *(const v8h*)(bp + 8);
    v16h b = __builtin_shufflevector(blo, bhi, 0, 1, 2, 3, 4, 5, 6, 7,
                                     8, 9, 10, 11, 12, 13, 14, 15);
    acc = __builtin_amdgcn_wmma_f32_16x16x32_f16(
        /*neg_a=*/false, a, /*neg_b=*/false, b,
        /*c_mod=*/(short)0, acc, /*reuse_a=*/false, /*reuse_b=*/false);
  }

#pragma unroll
  for (int r = 0; r < 8; ++r) {
    const int orow = tm * 16 + half * 8 + r;
    const int ocol = tn * 16 + l15;
    if (ocol < NREAL) C[(size_t)orow * NREAL + ocol] = acc[r];
  }
}

// ---------------- self-loop init (non-atomic) ------------------------------

__global__ void selfloop_h_k(const float* __restrict__ dinv,
                             const float* __restrict__ xw,
                             float* __restrict__ agg) {
  int idx = blockIdx.x * blockDim.x + threadIdx.x;
  if (idx < kN * kH) {
    int i = idx >> 7;  // / kH
    float w = dinv[i] * dinv[i];
    agg[idx] = w * xw[idx];
  }
}

__global__ void out_init_k(const float* __restrict__ dinv,
                           const float* __restrict__ xw2,
                           const float* __restrict__ b2,
                           float* __restrict__ out) {
  int idx = blockIdx.x * blockDim.x + threadIdx.x;
  if (idx < kN * kC) {
    int i = idx / kC;
    int c = idx - i * kC;
    float w = dinv[i] * dinv[i];
    out[idx] = w * xw2[idx] + b2[c];
  }
}

// ---------------- edge scatter (wave per edge) -----------------------------

__global__ void scatter_h_k(const int* __restrict__ src, const int* __restrict__ dst,
                            const float* __restrict__ dinv,
                            const float* __restrict__ xw,
                            float* __restrict__ agg) {
  const int wave = blockIdx.x * (blockDim.x >> 5) + (threadIdx.x >> 5);
  const int lane = threadIdx.x & 31;
  if (wave >= kE) return;
  const int s = src[wave];
  const int d = dst[wave];
  const float w = dinv[s] * dinv[d];
  const float4 v = ((const float4*)(xw + (size_t)s * kH))[lane];  // 512B/wave
  float* ag = agg + (size_t)d * kH + lane * 4;
  atomicAdd(ag + 0, w * v.x);
  atomicAdd(ag + 1, w * v.y);
  atomicAdd(ag + 2, w * v.z);
  atomicAdd(ag + 3, w * v.w);
}

__global__ void scatter_c_k(const int* __restrict__ src, const int* __restrict__ dst,
                            const float* __restrict__ dinv,
                            const float* __restrict__ xw2,
                            float* __restrict__ out) {
  const int wave = blockIdx.x * (blockDim.x >> 5) + (threadIdx.x >> 5);
  const int lane = threadIdx.x & 31;
  if (wave >= kE) return;
  const int s = src[wave];
  const int d = dst[wave];
  const float w = dinv[s] * dinv[d];
  for (int c = lane; c < kC; c += 32) {
    atomicAdd(out + (size_t)d * kC + c, w * xw2[(size_t)s * kC + c]);
  }
}

// ---------------- bias + ReLU ---------------------------------------------

__global__ void relu_bias_k(float* __restrict__ h, const float* __restrict__ b1) {
  int idx = blockIdx.x * blockDim.x + threadIdx.x;
  if (idx < kN * kH) {
    int f = idx & (kH - 1);
    float v = h[idx] + b1[f];
    h[idx] = v > 0.0f ? v : 0.0f;
  }
}

// ---------------- launch ---------------------------------------------------

extern "C" void kernel_launch(void* const* d_in, const int* in_sizes, int n_in,
                              void* d_out, int out_size, void* d_ws, size_t ws_size,
                              hipStream_t stream) {
  (void)in_sizes; (void)n_in; (void)out_size; (void)ws_size;

  const float* x   = (const float*)d_in[0];
  const int*   ei  = (const int*)d_in[1];
  const int*   src = ei;            // edge_index[0]
  const int*   dst = ei + kE;       // edge_index[1]
  const float* W1  = (const float*)d_in[2];
  const float* b1  = (const float*)d_in[3];
  const float* W2  = (const float*)d_in[4];
  const float* b2  = (const float*)d_in[5];
  float* out = (float*)d_out;

  // Workspace layout (floats): dinv[N] | xw[N*H] (reused for xw2) | agg[N*H]
  float* dinv = (float*)d_ws;
  float* xw   = dinv + kN;                 // byte offset 400000, 16B aligned
  float* agg  = xw + (size_t)kN * kH;

  const int T = 256;  // 8 waves per block for elementwise/scatter kernels

  // 1) degree (with self loops) -> D^{-1/2}
  deg_init_k<<<(kN + T - 1) / T, T, 0, stream>>>(dinv);
  deg_count_k<<<(kE + T - 1) / T, T, 0, stream>>>(dst, dinv);
  dinv_k<<<(kN + T - 1) / T, T, 0, stream>>>(dinv);

  // 2) xw1 = x @ W1   (K=100 padded to 128; 8 col tiles -> 256-thread blocks)
  wmma_gemm_k<kF, kH, 8><<<kN / 16, 8 * 32, 0, stream>>>(x, W1, xw);

  // 3) agg = selfloop + scatter-add over edges; h = relu(agg + b1)
  selfloop_h_k<<<(kN * kH + T - 1) / T, T, 0, stream>>>(dinv, xw, agg);
  scatter_h_k<<<(kE + 7) / 8, T, 0, stream>>>(src, dst, dinv, xw, agg);
  relu_bias_k<<<(kN * kH + T - 1) / T, T, 0, stream>>>(agg, b1);

  // 4) xw2 = h @ W2   (K=128 exact; 47 cols -> 3 col tiles, 96-thread blocks)
  wmma_gemm_k<kH, kC, 3><<<kN / 16, 3 * 32, 0, stream>>>(agg, W2, xw);

  // 5) out = selfloop + b2 + scatter-add over edges
  out_init_k<<<(kN * kC + T - 1) / T, T, 0, stream>>>(dinv, xw, b2, out);
  scatter_c_k<<<(kE + 7) / 8, T, 0, stream>>>(src, dst, dinv, xw, out);
}
